// KANLinearBSpline_58643483459979
// MI455X (gfx1250) — compile-verified, matrix operand order checked by reading, not run on hardware
//
#include <hip/hip_runtime.h>

// ---------------------------------------------------------------------------
// KANLinear forward on gfx1250 (MI455X), bf16 WMMA path, v5.
//   out[N][512] = SiLU(x) @ Wb^T + Bspline(x) @ (Ws * scaler)^T
// One GEMM, K = 512*8 (spline bases) + 512 (silu) = 4608.
//   - BN = 512 (full out width): each activation expanded exactly once.
//   - BM = 64, 512 threads (16 waves = 2M x 8N), 8 accumulators/wave.
//   - KT = 64, double-buffered LDS; B tiles staged via CDNA5 async LDS DMA
//     (GLOBAL_LOAD_ASYNC_TO_LDS_B128 + s_wait_asynccnt).
//   - Uniform-knot cubic B-spline via divided differences (FMA-only).
//   - v5: preload all B fragments per k-step so ds_load waits pipeline
//     incrementally instead of serializing each WMMA pair.
// ---------------------------------------------------------------------------

typedef __attribute__((ext_vector_type(16))) __bf16        v16bf;
typedef __attribute__((ext_vector_type(8)))  float         v8f;
typedef __attribute__((ext_vector_type(4)))  unsigned int  u32x4;
typedef __attribute__((ext_vector_type(4)))  int           v4i;
typedef __attribute__((ext_vector_type(4)))  float         f32x4;

#define IN_F   512
#define OUT_F  512
#define KSPL   4096            // 512 features * 8 spline bases
#define KTOT   4608            // + 512 silu features
#define BM     64              // C-tile rows per block
#define KT     64              // K-tile (2 WMMA k-steps of 32)
#define NKT    (KTOT / KT)     // 72 K-tiles (64 spline + 8 base)
#define LDP    72              // padded LDS row stride in bf16 (144B)
#define ASZ    (BM * LDP)      // shorts per A buffer
#define BSZ    (OUT_F * LDP)   // shorts per B buffer

#if __has_builtin(__builtin_amdgcn_global_load_async_to_lds_b128) && \
    __has_builtin(__builtin_amdgcn_s_wait_asynccnt)
#define USE_ASYNC_LDS 1
#else
#define USE_ASYNC_LDS 0
#endif

__device__ __forceinline__ unsigned short f2bf(float f) {
  unsigned int u = __builtin_bit_cast(unsigned int, f);
  u += 0x7FFFu + ((u >> 16) & 1u);            // round-to-nearest-even
  return (unsigned short)(u >> 16);
}

__device__ __forceinline__ unsigned int pack2bf(float a, float b) {
  unsigned int ua = __builtin_bit_cast(unsigned int, a);
  unsigned int ub = __builtin_bit_cast(unsigned int, b);
  ua += 0x7FFFu + ((ua >> 16) & 1u);
  ub += 0x7FFFu + ((ub >> 16) & 1u);
  return (ua >> 16) | (ub & 0xFFFF0000u);
}

__device__ __forceinline__ float silu_f(float x) {
  return x / (1.0f + __expf(-x));
}

#if USE_ASYNC_LDS
// Toolchain signature: (v4i addrspace(1)*, v4i addrspace(3)*, imm, imm)
__device__ __forceinline__ void async_b128(const void* g, void* l) {
  __builtin_amdgcn_global_load_async_to_lds_b128(
      (__attribute__((address_space(1))) v4i*)(void*)g,
      (__attribute__((address_space(3))) v4i*)l, 0, 0);
}
#endif

// 8 cubic B-spline bases on the uniform extended grid g[t] = (t-3)*1.2 - 3.
// Divided-difference form: basis m = (1/6) * sum_j (-1)^j C(4,j) relu(s-m-j)^3
// with s = (x+6.6)/1.2.  12 shared cubes + 8 FMA chains; no selects.
__device__ __forceinline__ void bsplines8_dd(float xv, float* o) {
  const float s = (xv + 6.6f) * (1.0f / 1.2f);
  float q[12];
#pragma unroll
  for (int k = 0; k < 12; ++k) {
    float t = fmaxf(s - (float)k, 0.0f);
    q[k] = t * t * t;
  }
#pragma unroll
  for (int m = 0; m < 8; ++m) {
    float v = q[m] - 4.0f * q[m + 1] + 6.0f * q[m + 2]
            - 4.0f * q[m + 3] + q[m + 4];
    o[m] = v * (1.0f / 6.0f);
  }
}

// ---------------------------------------------------------------------------
// Kernel 1: expand weights to bf16, K-major layout [OUT_F][KTOT].
// ---------------------------------------------------------------------------
__global__ void kan_expand_weights(const float* __restrict__ base_w,
                                   const float* __restrict__ spline_w,
                                   const float* __restrict__ scaler,
                                   unsigned short* __restrict__ wexp) {
  int idx = blockIdx.x * blockDim.x + threadIdx.x;   // over OUT_F*KTOT
  if (idx >= OUT_F * KTOT) return;
  int o = idx / KTOT;
  int k = idx - o * KTOT;
  float v;
  if (k < KSPL) {
    int i = k >> 3, bb = k & 7;
    v = spline_w[((size_t)o * IN_F + i) * 8 + bb] * scaler[(size_t)o * IN_F + i];
  } else {
    v = base_w[(size_t)o * IN_F + (k - KSPL)];
  }
  wexp[idx] = f2bf(v);
}

// ---------------------------------------------------------------------------
// Kernel 2: fused expansion + bf16 WMMA GEMM, double-buffered & pipelined.
// ---------------------------------------------------------------------------
__global__ __launch_bounds__(512)
void kan_wmma_gemm(const float* __restrict__ x,
                   const unsigned short* __restrict__ wexp,
                   float* __restrict__ out) {
  __shared__ __align__(16) unsigned short Alds[2 * ASZ];
  __shared__ __align__(16) unsigned short Blds[2 * BSZ];

  const int tid  = threadIdx.x;
  const int lane = tid & 31;
  const int wid  = tid >> 5;           // 0..15
  const int wm   = wid & 1;            // wave M: rows wm*32 .. +31
  const int wn   = wid >> 1;           // wave N: cols wn*64 .. +63
  const int half = lane >> 4;
  const int r    = lane & 15;

  const int m0 = blockIdx.x * BM;      // global token-row base

  v8f acc[2][4];
#pragma unroll
  for (int i = 0; i < 2; ++i)
#pragma unroll
    for (int j = 0; j < 4; ++j)
      acc[i][j] = v8f{};

  // A-staging registers (tile t+1 in flight while tile t computes)
  float xs = 0.0f;     // spline-region x sample
  f32x4 xb[2];         // base-region x chunks
#if !USE_ASYNC_LDS
  u32x4 breg[8];       // sync fallback: B tile via VGPRs
#endif

  // ---- B-tile staging -----------------------------------------------------
  auto stage_B = [&](int kt, int buf) {
    const int k0 = kt * KT;
    unsigned short* Bl = Blds + buf * BSZ;
#pragma unroll
    for (int j = 0; j < 8; ++j) {
      int c  = tid + j * 512;
      int n  = c >> 3;
      int kc = (c & 7) << 3;
#if USE_ASYNC_LDS
      // CDNA5 async LDS DMA: no VGPR data, tracked by ASYNCcnt
      async_b128(&wexp[(size_t)n * KTOT + k0 + kc], &Bl[n * LDP + kc]);
#else
      breg[j] = *(const u32x4*)(&wexp[(size_t)n * KTOT + k0 + kc]);
#endif
    }
  };

#if !USE_ASYNC_LDS
  auto store_B = [&](int buf) {
    unsigned short* Bl = Blds + buf * BSZ;
#pragma unroll
    for (int j = 0; j < 8; ++j) {
      int c  = tid + j * 512;
      int n  = c >> 3;
      int kc = (c & 7) << 3;
      *(u32x4*)(&Bl[n * LDP + kc]) = breg[j];
    }
  };
#endif

  // ---- A-tile staging (loads -> regs, then expand -> LDS) -----------------
  auto issue_A = [&](int kt) {
    const int k0 = kt * KT;
    if (k0 < KSPL) {                    // 8 features -> 1 sample / thread
      int fbase = k0 >> 3;
      int row = tid >> 3;
      xs = x[(size_t)(m0 + row) * IN_F + fbase + (tid & 7)];
    } else {                            // 64 silu features -> 2 f32x4 / thread
      int fb = k0 - KSPL;
#pragma unroll
      for (int j = 0; j < 2; ++j) {
        int c = tid + j * 512;
        int row = c >> 4, kc = (c & 15) << 2;
        xb[j] = *(const f32x4*)(&x[(size_t)(m0 + row) * IN_F + fb + kc]);
      }
    }
  };

  auto store_A = [&](int kt, int buf) {
    const int k0 = kt * KT;
    unsigned short* Al = Alds + buf * ASZ;
    if (k0 < KSPL) {
      int row = tid >> 3, fl = tid & 7;
      float bas[8];
      bsplines8_dd(xs, bas);
      u32x4 pk;
      pk.x = pack2bf(bas[0], bas[1]);
      pk.y = pack2bf(bas[2], bas[3]);
      pk.z = pack2bf(bas[4], bas[5]);
      pk.w = pack2bf(bas[6], bas[7]);
      *(u32x4*)(&Al[row * LDP + fl * 8]) = pk;
    } else {
#pragma unroll
      for (int j = 0; j < 2; ++j) {
        int c = tid + j * 512;
        int row = c >> 4, kc = (c & 15) << 2;
        unsigned long long p =
            (unsigned long long)pack2bf(silu_f(xb[j].x), silu_f(xb[j].y)) |
            ((unsigned long long)pack2bf(silu_f(xb[j].z), silu_f(xb[j].w)) << 32);
        *(unsigned long long*)(&Al[row * LDP + kc]) = p;
      }
    }
  };

  // ---- WMMA compute on one buffered K-tile --------------------------------
  auto compute = [&](int buf) {
    const unsigned short* Al = Alds + buf * ASZ;
    const unsigned short* Bl = Blds + buf * BSZ;
#pragma unroll
    for (int ks = 0; ks < 2; ++ks) {
      const int kk = ks * 32;
      // Preload ALL fragments for this k-step so the scheduler can pipeline
      // the dscnt waits (0x6 -> 0x4 -> 0x2 -> 0x0) across the WMMA burst.
      v16bf afr[2];
#pragma unroll
      for (int mi = 0; mi < 2; ++mi) {
        int mrow = wm * 32 + mi * 16 + r;
        union { u32x4 u[2]; v16bf v; } fa;
        // ISA 16-bit A layout: lanes 0-15 -> K {0..7,16..23}, lanes 16-31 -> +8
        fa.u[0] = *(const u32x4*)(&Al[mrow * LDP + kk + half * 8]);
        fa.u[1] = *(const u32x4*)(&Al[mrow * LDP + kk + 16 + half * 8]);
        afr[mi] = fa.v;
      }
      v16bf bfr[4];
#pragma unroll
      for (int ni = 0; ni < 4; ++ni) {
        int ncol = wn * 64 + ni * 16 + r;
        union { u32x4 u[2]; v16bf v; } fb;
        fb.u[0] = *(const u32x4*)(&Bl[ncol * LDP + kk + half * 8]);
        fb.u[1] = *(const u32x4*)(&Bl[ncol * LDP + kk + 16 + half * 8]);
        bfr[ni] = fb.v;
      }
#pragma unroll
      for (int ni = 0; ni < 4; ++ni)
#pragma unroll
        for (int mi = 0; mi < 2; ++mi)
          acc[mi][ni] = __builtin_amdgcn_wmma_f32_16x16x32_bf16(
              false, afr[mi], false, bfr[ni], (short)0, acc[mi][ni],
              false, false);
    }
  };

  // ---- prologue: stage tile 0 --------------------------------------------
  stage_B(0, 0);
  issue_A(0);
  store_A(0, 0);
#if USE_ASYNC_LDS
  __builtin_amdgcn_s_wait_asynccnt(0);
#else
  store_B(0);
#endif
  __syncthreads();

  // ---- main pipelined loop -----------------------------------------------
  for (int kt = 0; kt < NKT; ++kt) {
    const int cur = kt & 1;
    const int nxt = cur ^ 1;
    if (kt + 1 < NKT) {
      stage_B(kt + 1, nxt);   // async DMA in flight...
      issue_A(kt + 1);
    }
    compute(cur);             // ...while XDL chews on tile kt
    if (kt + 1 < NKT) {
      store_A(kt + 1, nxt);
#if !USE_ASYNC_LDS
      store_B(nxt);
#endif
    }
#if USE_ASYNC_LDS
    __builtin_amdgcn_s_wait_asynccnt(0);  // B DMA landed before barrier
#endif
    __syncthreads();
  }

  // ---- epilogue: C/D layout -> VGPR v holds row M = v + 8*half ------------
#pragma unroll
  for (int mi = 0; mi < 2; ++mi) {
#pragma unroll
    for (int v = 0; v < 8; ++v) {
      int row = m0 + wm * 32 + mi * 16 + v + 8 * half;
#pragma unroll
      for (int ni = 0; ni < 4; ++ni) {
        int col = wn * 64 + ni * 16 + r;
        out[(size_t)row * OUT_F + col] = acc[mi][ni][v];
      }
    }
  }
}

// ---------------------------------------------------------------------------
extern "C" void kernel_launch(void* const* d_in, const int* in_sizes, int n_in,
                              void* d_out, int out_size, void* d_ws, size_t ws_size,
                              hipStream_t stream) {
  const float* x        = (const float*)d_in[0];        // [N,512]
  const float* base_w   = (const float*)d_in[1];        // [512,512]
  const float* spline_w = (const float*)d_in[2];        // [512,512,8]
  const float* scaler   = (const float*)d_in[3];        // [512,512]
  // d_in[4] = grid: uniform, recomputed as constants in-kernel
  float* out = (float*)d_out;

  unsigned short* wexp = (unsigned short*)d_ws;          // 512*4608*2B = 4.7MB

  const int nrows = in_sizes[0] / IN_F;                  // 32768

  int nw = OUT_F * KTOT;
  kan_expand_weights<<<(nw + 255) / 256, 256, 0, stream>>>(base_w, spline_w,
                                                           scaler, wexp);

  kan_wmma_gemm<<<nrows / BM, 512, 0, stream>>>(x, wexp, out);
}
